// GeneModelClassic_64768106824288
// MI455X (gfx1250) — matrix operand order repaired and checked by reading, not compiled, not run on hardware
//
#include <hip/hip_runtime.h>

// Problem constants (from the reference)
#define B_DIM 32
#define G_DIM 18000
#define K_DIM 50
#define L_DIM 16
#define N_SNPS (G_DIM * K_DIM)      // 900000
#define GENES_PER_BLOCK 8
#define K_PAD 64                     // 50 -> 64, two WMMA k-steps of 32
#define XS_STRIDE (GENES_PER_BLOCK * K_PAD + 8)  // 520 halfs: bank-conflict-free row stride
#define WT_STRIDE (K_PAD + 8)                    // 72 halfs

typedef __attribute__((ext_vector_type(16))) _Float16 v16h;
typedef __attribute__((ext_vector_type(8)))  _Float16 v8h;
typedef __attribute__((ext_vector_type(8)))  float    v8f;

__global__ __launch_bounds__(256) void gene_wmma_kernel(
    const float* __restrict__ x,        // (B, N_SNPS)
    const float* __restrict__ scale_w,  // (N_SNPS)
    const float* __restrict__ scale_b,  // (N_SNPS)
    const float* __restrict__ W,        // (G, K, L) row-major
    const float* __restrict__ bias,     // (G, L)
    const int*   __restrict__ snp_index,// (G, K)
    float*       __restrict__ out)      // (B, G*L)
{
    __shared__ __align__(16) _Float16 xs_tile[B_DIM * XS_STRIDE];                      // 33280 B
    __shared__ __align__(16) _Float16 wt_tile[GENES_PER_BLOCK * L_DIM * WT_STRIDE];    // 18432 B

    const int tid = threadIdx.x;
    const int g0  = blockIdx.x * GENES_PER_BLOCK;

    // ---- zero the K padding (50..63) so WMMA over K=64 is exact ----
    for (int i = tid; i < B_DIM * GENES_PER_BLOCK * (K_PAD - K_DIM); i += 256) {
        const int b  = i / (GENES_PER_BLOCK * (K_PAD - K_DIM));
        const int r  = i % (GENES_PER_BLOCK * (K_PAD - K_DIM));
        const int gl = r / (K_PAD - K_DIM);
        const int k  = K_DIM + r % (K_PAD - K_DIM);
        xs_tile[b * XS_STRIDE + gl * K_PAD + k] = (_Float16)0.f;
    }
    for (int i = tid; i < GENES_PER_BLOCK * L_DIM * (K_PAD - K_DIM); i += 256) {
        const int gl = i / (L_DIM * (K_PAD - K_DIM));
        const int r  = i % (L_DIM * (K_PAD - K_DIM));
        const int n  = r / (K_PAD - K_DIM);
        const int k  = K_DIM + r % (K_PAD - K_DIM);
        wt_tile[(gl * L_DIM + n) * WT_STRIDE + k] = (_Float16)0.f;
    }

    // ---- per-thread SNP metadata (400 slots over 256 threads) ----
    const int nslots = GENES_PER_BLOCK * K_DIM;   // 400
    int   c0 = 0, c1 = 0, o0 = 0, o1 = 0;
    float w0 = 0.f, s0 = 0.f, w1 = 0.f, s1 = 0.f;
    if (tid < nslots) {
        c0 = snp_index[g0 * K_DIM + tid];
        w0 = scale_w[c0];
        s0 = scale_b[c0];
        o0 = (tid / K_DIM) * K_PAD + (tid % K_DIM);
    }
    const int t1 = tid + 256;
    if (t1 < nslots) {
        c1 = snp_index[g0 * K_DIM + t1];
        w1 = scale_w[c1];
        s1 = scale_b[c1];
        o1 = (t1 / K_DIM) * K_PAD + (t1 % K_DIM);
    }

    // ---- stage xs = x*scale_w + scale_b, gathered, f16, coalesced & streaming ----
    for (int b = 0; b < B_DIM; ++b) {
        const float* xrow = x + (size_t)b * N_SNPS;
        if (tid < nslots)
            xs_tile[b * XS_STRIDE + o0] =
                (_Float16)(__builtin_nontemporal_load(xrow + c0) * w0 + s0);
        if (t1 < nslots)
            xs_tile[b * XS_STRIDE + o1] =
                (_Float16)(__builtin_nontemporal_load(xrow + c1) * w1 + s1);
    }

    // ---- stage W transposed to [gene][N][K] f16 (coalesced global reads) ----
    const float* Wg = W + (size_t)g0 * (K_DIM * L_DIM);
    for (int i = tid; i < GENES_PER_BLOCK * K_DIM * L_DIM; i += 256) {
        const float wv = __builtin_nontemporal_load(Wg + i);
        const int gl = i / (K_DIM * L_DIM);
        const int r  = i % (K_DIM * L_DIM);
        const int k  = r >> 4;      // K index
        const int n  = r & 15;      // L index
        wt_tile[(gl * L_DIM + n) * WT_STRIDE + k] = (_Float16)wv;
    }

    __syncthreads();

    // ---- one wave == one gene: (32x64) x (64x16) via 2 M-tiles x 2 k-steps ----
    const int wave = tid >> 5;        // gene within block
    const int lane = tid & 31;
    const int g    = g0 + wave;
    const int mrow = lane & 15;       // A row (M) and B/C column (N)
    const int hi16 = lane >> 4;       // lane half selector

    v8f acc0 = {};  // batch rows 0..15
    v8f acc1 = {};  // batch rows 16..31

#pragma unroll
    for (int ks = 0; ks < 2; ++ks) {
        // A-fragment (16-bit 16x32): lane L<16 holds K runs [koff,koff+8) & [koff+16,koff+24)
        const int akoff = wave * K_PAD + ks * 32 + 8 * hi16;
        const _Float16* a0p = &xs_tile[mrow * XS_STRIDE + akoff];
        const _Float16* a1p = &xs_tile[(16 + mrow) * XS_STRIDE + akoff];
        const v8h a0lo = *(const v8h*)(a0p);
        const v8h a0hi = *(const v8h*)(a0p + 16);
        const v8h a1lo = *(const v8h*)(a1p);
        const v8h a1hi = *(const v8h*)(a1p + 16);

        // B-fragment (16-bit 32x16): lane L<16 holds K runs [koff,koff+8) & [koff+16,koff+24) of column N=L
        const _Float16* bp = &wt_tile[(wave * L_DIM + mrow) * WT_STRIDE + ks * 32 + 8 * hi16];
        const v8h blo = *(const v8h*)(bp);
        const v8h bhi = *(const v8h*)(bp + 16);

        const v16h av0 = __builtin_shufflevector(a0lo, a0hi, 0,1,2,3,4,5,6,7,8,9,10,11,12,13,14,15);
        const v16h av1 = __builtin_shufflevector(a1lo, a1hi, 0,1,2,3,4,5,6,7,8,9,10,11,12,13,14,15);
        const v16h bv  = __builtin_shufflevector(blo,  bhi,  0,1,2,3,4,5,6,7,8,9,10,11,12,13,14,15);

        acc0 = __builtin_amdgcn_wmma_f32_16x16x32_f16(false, av0, false, bv, (short)0, acc0, false, false);
        acc1 = __builtin_amdgcn_wmma_f32_16x16x32_f16(false, av1, false, bv, (short)0, acc1, false, false);
    }

    // ---- bias + ReLU + coalesced nontemporal stores ----
    const float bv = bias[(size_t)g * L_DIM + mrow];
    float* outg = out + (size_t)g * L_DIM + mrow;
#pragma unroll
    for (int r = 0; r < 8; ++r) {
        const int m0 = 8 * hi16 + r;        // C/D layout: VGPR r -> M = 8*(lane/16)+r
        float v0 = acc0[r] + bv; v0 = v0 > 0.f ? v0 : 0.f;
        __builtin_nontemporal_store(v0, outg + (size_t)m0 * (G_DIM * L_DIM));
        const int m1 = 16 + m0;
        float v1 = acc1[r] + bv; v1 = v1 > 0.f ? v1 : 0.f;
        __builtin_nontemporal_store(v1, outg + (size_t)m1 * (G_DIM * L_DIM));
    }
}

extern "C" void kernel_launch(void* const* d_in, const int* in_sizes, int n_in,
                              void* d_out, int out_size, void* d_ws, size_t ws_size,
                              hipStream_t stream) {
    (void)in_sizes; (void)n_in; (void)out_size; (void)d_ws; (void)ws_size;
    const float* x   = (const float*)d_in[0];
    const float* sw  = (const float*)d_in[1];
    const float* sb  = (const float*)d_in[2];
    const float* W   = (const float*)d_in[3];
    const float* bia = (const float*)d_in[4];
    const int*   idx = (const int*)d_in[5];
    float* out = (float*)d_out;

    dim3 grid(G_DIM / GENES_PER_BLOCK);   // 2250 blocks
    dim3 block(256);                      // 8 wave32 -> 8 genes per block
    gene_wmma_kernel<<<grid, block, 0, stream>>>(x, sw, sb, W, bia, idx, out);
}
